// RingAttention_4964982194278
// MI455X (gfx1250) — matrix operand order repaired and checked by reading, not compiled, not run on hardware
//
#include <hip/hip_runtime.h>
#include <hip/hip_bf16.h>

typedef __bf16 bf16_t;
typedef __bf16 bf16x4 __attribute__((ext_vector_type(4)));
typedef __bf16 bf16x8 __attribute__((ext_vector_type(8)));
typedef __bf16 v16bf  __attribute__((ext_vector_type(16)));
typedef float  v8f    __attribute__((ext_vector_type(8)));
typedef unsigned int v4u __attribute__((ext_vector_type(4)));
typedef int          v4i __attribute__((ext_vector_type(4)));
typedef int          v8i __attribute__((ext_vector_type(8)));

#define SEQ     4096
#define HIDDEN  1024
#define HEADS   16
#define HDIM    64
#define KVBLK   1024

// ---- WMMA fragment helpers (bf16, 16x16x32, f32 accum) ----
// A frag (16x32, lane = row, half=lane>>4): k = half*16+0..15 (contiguous)
// B frag (32x16, lane = col, half=lane>>4): k = half*16+0..15 (contiguous)
static __device__ __forceinline__ v16bf frag_cat(bf16x8 lo, bf16x8 hi) {
  return __builtin_shufflevector(lo, hi, 0,1,2,3,4,5,6,7,8,9,10,11,12,13,14,15);
}
static __device__ __forceinline__ v16bf load_a_frag(const bf16_t* p) {
  return frag_cat(*(const bf16x8*)p, *(const bf16x8*)(p + 8));
}
static __device__ __forceinline__ v16bf load_b_frag(const bf16_t* p) {
  return frag_cat(*(const bf16x8*)p, *(const bf16x8*)(p + 8));
}
static __device__ __forceinline__ v8f wmma_bf16(v16bf a, v16bf b, v8f c) {
  return __builtin_amdgcn_wmma_f32_16x16x32_bf16(false, a, false, b, (short)0, c,
                                                 false, false);
}
static __device__ __forceinline__ v8f v8f_zero() {
  v8f z = {0.f,0.f,0.f,0.f,0.f,0.f,0.f,0.f};
  return z;
}

// ---- LDS byte-offset of a __shared__ object (generic -> AS3 -> int) ----
static __device__ __forceinline__ unsigned lds_offset_of(const void* p) {
  return (unsigned)(unsigned long long)(__attribute__((address_space(3))) const char*)p;
}

// ---- TDM: async-load a 32x64 bf16 tile (row stride 64 elems) into LDS ----
// D# per CDNA5 ISA: group0 = {count|flags, lds_addr, gaddr_lo, gaddr_hi|type=2}
// group1 = dims/strides; groups 2-3 zero (2D tensor). 6-arg builtin form.
static __device__ __forceinline__ void tdm_load_k_tile(const bf16_t* gsrc,
                                                       unsigned lds_byte_off) {
  unsigned long long ga = (unsigned long long)gsrc;
  v4u g0;
  g0[0] = 1u;                                         // count=1, user descriptor
  g0[1] = lds_byte_off;                               // lds_addr
  g0[2] = (unsigned)(ga & 0xffffffffu);               // global_addr[31:0]
  g0[3] = (unsigned)((ga >> 32) & 0x01ffffffu)        // global_addr[56:32]
        | 0x80000000u;                                // type = 2 ("image")
  v8i g1;
  g1[0] = (int)(1u << 16);          // data_size=1 (2 bytes), no multicast/pad
  g1[1] = (int)(64u << 16);         // tensor_dim0 = 64 (bits 63:48)
  g1[2] = (int)(4096u << 16);       // dim0 hi=0 | tensor_dim1 = 4096 (lo16)
  g1[3] = (int)(64u << 16);         // dim1 hi=0 | tile_dim0 = 64
  g1[4] = 32;                       // tile_dim1 = 32, tile_dim2 = 0
  g1[5] = 64;                       // tensor_dim0_stride = 64 (lo32)
  g1[6] = 0;                        // stride0 hi | tensor_dim1_stride lo
  g1[7] = 0;
  v4i z4 = {0,0,0,0};
  v8i z8 = {0,0,0,0,0,0,0,0};
  __builtin_amdgcn_tensor_load_to_lds(g0, g1, z4, z4, z8, 0);
}

// ---- fp32 -> bf16 bulk convert (4 elems/thread) ----
__global__ void cvt_f32_bf16(const float* __restrict__ in, bf16_t* __restrict__ out, int n4) {
  int i = blockIdx.x * blockDim.x + threadIdx.x;
  if (i < n4) {
    float4 v = ((const float4*)in)[i];
    bf16x4 o = { (bf16_t)v.x, (bf16_t)v.y, (bf16_t)v.z, (bf16_t)v.w };
    ((bf16x4*)out)[i] = o;
  }
}

// ---- 1024x1024 fp32 -> transposed bf16 [out][in], LDS tiled ----
__global__ void transpose_cvt(const float* __restrict__ w, bf16_t* __restrict__ wt) {
  __shared__ float tile[32][33];
  int tx = threadIdx.x, ty = threadIdx.y;          // 32 x 8
  int x0 = blockIdx.x * 32, y0 = blockIdx.y * 32;
  #pragma unroll
  for (int i = 0; i < 32; i += 8)
    tile[ty + i][tx] = w[(size_t)(y0 + ty + i) * HIDDEN + x0 + tx];
  __syncthreads();
  #pragma unroll
  for (int i = 0; i < 32; i += 8)
    wt[(size_t)(x0 + ty + i) * HIDDEN + y0 + tx] = (bf16_t)tile[tx][ty + i];
}

// ---- GEMM: C[4096x1024] = A[4096x1024] * Bt^T, bf16 in, f32 accum ----
// Block = 8 waves as 2x4; wave tile = 32x64 (2 A-frags x 4 B-frags, 8 WMMA/step).
// Modulo-2 software pipeline with named alternating buffers; loop kept ROLLED
// (unroll 1) so the scheduler preserves the one-stage prefetch distance; final
// iteration peeled so the hot loop has no bounds guard.
// mode 0: bf16 out [head][s][d]   (Q, K)
// mode 1: bf16 out [col][row]     (V -> [h][d][s])
// mode 2: f32  out row-major      (final output)
__global__ __launch_bounds__(256) void gemm_bf16_wmma(
    const bf16_t* __restrict__ A, const bf16_t* __restrict__ Bt,
    void* __restrict__ outp, float scale, int mode) {
  const int lane = threadIdx.x & 31;
  const int wave = threadIdx.x >> 5;
  const int hlf  = lane >> 4;
  const int l16  = lane & 15;
  const int wr   = wave >> 2;        // 0..1
  const int wc   = wave & 3;         // 0..3
  const int m0 = blockIdx.y * 64 + wr * 32;
  const int n0 = blockIdx.x * 256 + wc * 64;

  v8f acc[2][4];
  #pragma unroll
  for (int mi = 0; mi < 2; ++mi)
    #pragma unroll
    for (int j = 0; j < 4; ++j) acc[mi][j] = v8f_zero();

  const bf16_t* aRow[2];
  aRow[0] = A + (size_t)(m0 + l16) * HIDDEN + hlf * 16;
  aRow[1] = aRow[0] + 16 * HIDDEN;
  const bf16_t* bRow[4];
  #pragma unroll
  for (int j = 0; j < 4; ++j)
    bRow[j] = Bt + (size_t)(n0 + j * 16 + l16) * HIDDEN + hlf * 16;

  v16bf aP[2][2], bP[2][4];          // [buffer][frag]
  #pragma unroll
  for (int mi = 0; mi < 2; ++mi) aP[0][mi] = load_a_frag(aRow[mi]);
  #pragma unroll
  for (int j = 0; j < 4; ++j) bP[0][j] = load_b_frag(bRow[j]);

  #pragma unroll 1
  for (int k0 = 0; k0 < HIDDEN - 64; k0 += 64) {
    // stage A: prefetch k0+32 into buf1, compute on buf0
    #pragma unroll
    for (int mi = 0; mi < 2; ++mi) aP[1][mi] = load_a_frag(aRow[mi] + k0 + 32);
    #pragma unroll
    for (int j = 0; j < 4; ++j) bP[1][j] = load_b_frag(bRow[j] + k0 + 32);
    #pragma unroll
    for (int j = 0; j < 4; ++j) {
      acc[0][j] = wmma_bf16(aP[0][0], bP[0][j], acc[0][j]);
      acc[1][j] = wmma_bf16(aP[0][1], bP[0][j], acc[1][j]);
    }
    // stage B: prefetch k0+64 into buf0, compute on buf1
    #pragma unroll
    for (int mi = 0; mi < 2; ++mi) aP[0][mi] = load_a_frag(aRow[mi] + k0 + 64);
    #pragma unroll
    for (int j = 0; j < 4; ++j) bP[0][j] = load_b_frag(bRow[j] + k0 + 64);
    #pragma unroll
    for (int j = 0; j < 4; ++j) {
      acc[0][j] = wmma_bf16(aP[1][0], bP[1][j], acc[0][j]);
      acc[1][j] = wmma_bf16(aP[1][1], bP[1][j], acc[1][j]);
    }
  }
  // peeled final 64: k = HIDDEN-64 (buf0 already loaded) and HIDDEN-32 (buf1)
  {
    const int kt = HIDDEN - 32;
    #pragma unroll
    for (int mi = 0; mi < 2; ++mi) aP[1][mi] = load_a_frag(aRow[mi] + kt);
    #pragma unroll
    for (int j = 0; j < 4; ++j) bP[1][j] = load_b_frag(bRow[j] + kt);
    #pragma unroll
    for (int j = 0; j < 4; ++j) {
      acc[0][j] = wmma_bf16(aP[0][0], bP[0][j], acc[0][j]);
      acc[1][j] = wmma_bf16(aP[0][1], bP[0][j], acc[1][j]);
    }
    #pragma unroll
    for (int j = 0; j < 4; ++j) {
      acc[0][j] = wmma_bf16(aP[1][0], bP[1][j], acc[0][j]);
      acc[1][j] = wmma_bf16(aP[1][1], bP[1][j], acc[1][j]);
    }
  }

  #pragma unroll
  for (int mi = 0; mi < 2; ++mi) {
    #pragma unroll
    for (int j = 0; j < 4; ++j) {
      int col = n0 + j * 16 + l16;
      #pragma unroll
      for (int r = 0; r < 8; ++r) {
        int row = m0 + mi * 16 + hlf * 8 + r;
        float v = acc[mi][j][r] * scale;
        if (mode == 0) {
          ((bf16_t*)outp)[(size_t)(col >> 6) * (SEQ * HDIM) + (size_t)row * HDIM + (col & 63)] = (bf16_t)v;
        } else if (mode == 1) {
          ((bf16_t*)outp)[(size_t)col * SEQ + row] = (bf16_t)v;
        } else {
          ((float*)outp)[(size_t)row * HIDDEN + col] = v;
        }
      }
    }
  }
}

// ---- Ring attention, faithful block-local-max scheme ----
// Q,K: [h][s][64] bf16 (Q pre-scaled by 1/8). Vt: [h][64][s] bf16. O: [s][1024] bf16.
// K tiles (32x64) staged to LDS by the Tensor Data Mover, double-buffered.
__global__ __launch_bounds__(256) void attn_wmma(
    const bf16_t* __restrict__ Q, const bf16_t* __restrict__ K,
    const bf16_t* __restrict__ Vt, bf16_t* __restrict__ O) {
  struct __align__(16) AttnSmem {
    bf16_t kt[2][32][64];    // double-buffered K tile (TDM destination)
    bf16_t et[8][16][40];    // per-wave E-tile exchange (C-layout -> A-layout)
  };
  __shared__ AttnSmem sm;

  const int lane = threadIdx.x & 31;
  const int wave = threadIdx.x >> 5;
  const int hlf  = lane >> 4;
  const int l16  = lane & 15;
  const int h  = blockIdx.y;
  const int q0 = blockIdx.x * 128 + wave * 16;

  const bf16_t* Qh = Q  + (size_t)h * (SEQ * HDIM);
  const bf16_t* Kh = K  + (size_t)h * (SEQ * HDIM);
  const bf16_t* Vh = Vt + (size_t)h * (HDIM * SEQ);

  unsigned ktOff[2];
  ktOff[0] = lds_offset_of(&sm.kt[0][0][0]);
  ktOff[1] = lds_offset_of(&sm.kt[1][0][0]);

  const bf16_t* qRow = Qh + (size_t)(q0 + l16) * HDIM;
  v16bf aQ0 = load_a_frag(qRow + hlf * 16);        // d = 0..31
  v16bf aQ1 = load_a_frag(qRow + 32 + hlf * 16);   // d = 32..63

  v8f num[4];
  #pragma unroll
  for (int j = 0; j < 4; ++j) num[j] = v8f_zero();
  float dsum[8];
  #pragma unroll
  for (int r = 0; r < 8; ++r) dsum[r] = 0.f;

  const int NT = KVBLK / 32;   // 32 K-tiles of 32 rows per KV block

  for (int b = 0; b < SEQ / KVBLK; ++b) {
    const int kv0 = b * KVBLK;

    // ---------------- pass 1: block-local row max ----------------
    if (wave == 0) tdm_load_k_tile(Kh + (size_t)kv0 * HDIM, ktOff[0]);
    float pm[8];
    #pragma unroll
    for (int r = 0; r < 8; ++r) pm[r] = -3.402823466e38f;

    for (int t = 0; t < NT; ++t) {
      if (wave == 0) {
        if (t + 1 < NT) {
          tdm_load_k_tile(Kh + (size_t)(kv0 + (t + 1) * 32) * HDIM, ktOff[(t + 1) & 1]);
          __builtin_amdgcn_s_wait_tensorcnt(1);   // tile t complete
        } else {
          __builtin_amdgcn_s_wait_tensorcnt(0);
        }
      }
      __syncthreads();                            // publish tile t
      const bf16_t (*kt)[64] = sm.kt[t & 1];
      #pragma unroll
      for (int jj = 0; jj < 2; ++jj) {
        const bf16_t* kr = kt[jj * 16 + l16];
        v8f s = v8f_zero();
        s = wmma_bf16(aQ0, load_b_frag(kr + hlf * 16), s);
        s = wmma_bf16(aQ1, load_b_frag(kr + 32 + hlf * 16), s);
        #pragma unroll
        for (int r = 0; r < 8; ++r) pm[r] = fmaxf(pm[r], s[r]);
      }
      __syncthreads();                            // reads done, buffer reusable
    }
    #pragma unroll
    for (int r = 0; r < 8; ++r) {
      pm[r] = fmaxf(pm[r], __shfl_xor(pm[r], 1, 32));
      pm[r] = fmaxf(pm[r], __shfl_xor(pm[r], 2, 32));
      pm[r] = fmaxf(pm[r], __shfl_xor(pm[r], 4, 32));
      pm[r] = fmaxf(pm[r], __shfl_xor(pm[r], 8, 32));
    }

    // ------- pass 2: e = exp(s - m); num += E*V, den += sum(e) -------
    if (wave == 0) tdm_load_k_tile(Kh + (size_t)kv0 * HDIM, ktOff[0]);
    for (int t = 0; t < NT; ++t) {
      if (wave == 0) {
        if (t + 1 < NT) {
          tdm_load_k_tile(Kh + (size_t)(kv0 + (t + 1) * 32) * HDIM, ktOff[(t + 1) & 1]);
          __builtin_amdgcn_s_wait_tensorcnt(1);
        } else {
          __builtin_amdgcn_s_wait_tensorcnt(0);
        }
      }
      __syncthreads();
      const bf16_t (*kt)[64] = sm.kt[t & 1];
      #pragma unroll
      for (int jj = 0; jj < 2; ++jj) {
        const bf16_t* kr = kt[jj * 16 + l16];
        v8f s = v8f_zero();
        s = wmma_bf16(aQ0, load_b_frag(kr + hlf * 16), s);
        s = wmma_bf16(aQ1, load_b_frag(kr + 32 + hlf * 16), s);
        #pragma unroll
        for (int r = 0; r < 8; ++r) {
          float e = __expf(s[r] - pm[r]);
          dsum[r] += e;
          sm.et[wave][hlf * 8 + r][jj * 16 + l16] = (bf16_t)e;  // C-layout -> LDS
        }
      }
      asm volatile("s_wait_dscnt 0" ::: "memory");               // intra-wave exchange
      v16bf aE = load_a_frag(&sm.et[wave][l16][hlf * 16]);        // A-layout read back
      v16bf vF[4];
      #pragma unroll
      for (int jd = 0; jd < 4; ++jd)
        vF[jd] = load_b_frag(Vh + (size_t)(jd * 16 + l16) * SEQ + kv0 + t * 32 + hlf * 16);
      #pragma unroll
      for (int jd = 0; jd < 4; ++jd)
        num[jd] = wmma_bf16(aE, vF[jd], num[jd]);
      __syncthreads();
    }
  }

  #pragma unroll
  for (int r = 0; r < 8; ++r) {
    dsum[r] += __shfl_xor(dsum[r], 1, 32);
    dsum[r] += __shfl_xor(dsum[r], 2, 32);
    dsum[r] += __shfl_xor(dsum[r], 4, 32);
    dsum[r] += __shfl_xor(dsum[r], 8, 32);
  }
  #pragma unroll
  for (int jd = 0; jd < 4; ++jd) {
    int col = h * HDIM + jd * 16 + l16;
    #pragma unroll
    for (int r = 0; r < 8; ++r)
      O[(size_t)(q0 + hlf * 8 + r) * HIDDEN + col] = (bf16_t)(num[jd][r] / dsum[r]);
  }
}

extern "C" void kernel_launch(void* const* d_in, const int* in_sizes, int n_in,
                              void* d_out, int out_size, void* d_ws, size_t ws_size,
                              hipStream_t stream) {
  (void)in_sizes; (void)n_in; (void)out_size; (void)ws_size;
  const float* hs = (const float*)d_in[0];
  const float* wq = (const float*)d_in[1];
  const float* wk = (const float*)d_in[2];
  const float* wv = (const float*)d_in[3];
  const float* wo = (const float*)d_in[4];

  char* ws = (char*)d_ws;
  size_t off = 0;
  bf16_t* hb  = (bf16_t*)(ws + off); off += (size_t)SEQ * HIDDEN * 2;
  bf16_t* wqT = (bf16_t*)(ws + off); off += (size_t)HIDDEN * HIDDEN * 2;
  bf16_t* wkT = (bf16_t*)(ws + off); off += (size_t)HIDDEN * HIDDEN * 2;
  bf16_t* wvT = (bf16_t*)(ws + off); off += (size_t)HIDDEN * HIDDEN * 2;
  bf16_t* woT = (bf16_t*)(ws + off); off += (size_t)HIDDEN * HIDDEN * 2;
  bf16_t* Qb  = (bf16_t*)(ws + off); off += (size_t)SEQ * HIDDEN * 2;  // [h][s][d], pre-scaled
  bf16_t* Kb  = (bf16_t*)(ws + off); off += (size_t)SEQ * HIDDEN * 2;  // [h][s][d]
  bf16_t* Vtb = (bf16_t*)(ws + off); off += (size_t)SEQ * HIDDEN * 2;  // [h][d][s]
  bf16_t* Ob  = (bf16_t*)(ws + off); off += (size_t)SEQ * HIDDEN * 2;  // [s][1024]

  int n4 = SEQ * HIDDEN / 4;
  cvt_f32_bf16<<<(n4 + 255) / 256, 256, 0, stream>>>(hs, hb, n4);

  dim3 tb(32, 8), tg(32, 32);
  transpose_cvt<<<tg, tb, 0, stream>>>(wq, wqT);
  transpose_cvt<<<tg, tb, 0, stream>>>(wk, wkT);
  transpose_cvt<<<tg, tb, 0, stream>>>(wv, wvT);
  transpose_cvt<<<tg, tb, 0, stream>>>(wo, woT);

  dim3 gg(HIDDEN / 256, SEQ / 64);  // (4, 64)
  gemm_bf16_wmma<<<gg, 256, 0, stream>>>(hb, wqT, Qb, 0.125f, 0);  // scale folded into Q
  gemm_bf16_wmma<<<gg, 256, 0, stream>>>(hb, wkT, Kb, 1.0f, 0);
  gemm_bf16_wmma<<<gg, 256, 0, stream>>>(hb, wvT, Vtb, 1.0f, 1);

  attn_wmma<<<dim3(SEQ / 128, HEADS), 256, 0, stream>>>(Qb, Kb, Vtb, Ob);

  gemm_bf16_wmma<<<gg, 256, 0, stream>>>(Ob, woT, (void*)d_out, 1.0f, 2);
}
